// ZeroshotProximityModel_61375082660556
// MI455X (gfx1250) — matrix operand order repaired and checked by reading, not compiled, not run on hardware
//
#include <hip/hip_runtime.h>
#include <hip/hip_fp16.h>
#include <stdint.h>

// Problem constants (from reference setup_inputs)
#define B_DIM 32
#define L_DIM 1024
#define K_DIM 4096
#define D_DIM 512
#define GAP   (-10)

typedef _Float16 v16h __attribute__((ext_vector_type(16)));
typedef _Float16 v8h  __attribute__((ext_vector_type(8)));
typedef float    v8f  __attribute__((ext_vector_type(8)));

// ---------------------------------------------------------------------------
// Kernel 1: L2-normalize codebook rows (fp32) and cast to fp16.
// One 256-thread block per row (4096 rows x 512 floats).
// ---------------------------------------------------------------------------
__global__ __launch_bounds__(256) void normalize_embed(
    const float* __restrict__ embed, _Float16* __restrict__ E) {
  const int row = blockIdx.x;
  const int tid = threadIdx.x;
  const int lane = tid & 31;
  const int wv = tid >> 5;
  const float* src = embed + (size_t)row * D_DIM;

  float ss = 0.f;
  for (int k = tid; k < D_DIM; k += 256) {
    float v = src[k];
    ss += v * v;
  }
#pragma unroll
  for (int d = 16; d >= 1; d >>= 1) ss += __shfl_xor(ss, d, 32);

  __shared__ float part[8];
  if (lane == 0) part[wv] = ss;
  __syncthreads();
  float tot = 0.f;
#pragma unroll
  for (int i = 0; i < 8; ++i) tot += part[i];
  const float norm = sqrtf(tot);

  for (int k = tid; k < D_DIM; k += 256)
    E[(size_t)row * D_DIM + k] = (_Float16)(src[k] / norm);
}

// ---------------------------------------------------------------------------
// Kernel 2: sub = int8( fp16( E @ E^T ) * 100 )  via v_wmma_f32_16x16x32_f16.
// One wave computes a 32x32 tile of the 4096x4096 output (2x2 fragments).
// ---------------------------------------------------------------------------

// A-matrix fragment, 16x32 f16 (CDNA5 ISA 16-bit A layout):
//   lane 0-15:  row = lane,   VGPR0-3 hold K=0..7,   VGPR4-7 hold K=16..23
//   lane 16-31: row = lane-16,VGPR0-3 hold K=8..15,  VGPR4-7 hold K=24..31
__device__ inline v16h load_a_frag(const _Float16* __restrict__ E,
                                   int rowBase, int kb, int lane) {
  const int r = rowBase + (lane & 15);
  const int koff = (lane >> 4) << 3;  // 0 or 8
  const _Float16* p = E + (size_t)r * D_DIM + kb + koff;
  v8h lo = *(const v8h*)(p);         // K = kb+koff   .. +7
  v8h hi = *(const v8h*)(p + 16);    // K = kb+16+koff.. +7
  v16h f;
#pragma unroll
  for (int i = 0; i < 8; ++i) { f[i] = lo[i]; f[i + 8] = hi[i]; }
  return f;
}

// B-matrix fragment, 32x16 f16: B = E^T so column n of B is row n of E.
//   lane 0-15:  col = lane,    VGPR0-3 hold K = kb+0..7,  VGPR4-7 hold K = kb+16..23
//   lane 16-31: col = lane-16, VGPR0-3 hold K = kb+8..15, VGPR4-7 hold K = kb+24..31
__device__ inline v16h load_b_frag(const _Float16* __restrict__ E,
                                   int colBase, int kb, int lane) {
  const int r = colBase + (lane & 15);
  const int koff = (lane >> 4) << 3;  // 0 or 8
  const _Float16* p = E + (size_t)r * D_DIM + kb + koff;
  v8h lo = *(const v8h*)(p);
  v8h hi = *(const v8h*)(p + 16);
  v16h f;
#pragma unroll
  for (int i = 0; i < 8; ++i) { f[i] = lo[i]; f[i + 8] = hi[i]; }
  return f;
}

// C/D layout: lane l, VGPR r -> (m = r + 8*(l>=16), n = l&15).
// Convert f32 acc -> fp16 (RNE), * fp16(100), truncate -> int8 (matches
// reference's fp16 matmul result * fp16(100) astype(int32)).
__device__ inline void store_tile(int8_t* __restrict__ sub, const v8f& c,
                                  int mBase, int nBase, int mo, int nl) {
#pragma unroll
  for (int r = 0; r < 8; ++r) {
    _Float16 h = (_Float16)c[r];
    h = h * (_Float16)100.0f;
    int iv = (int)h;  // C truncation toward zero, same as astype(int32)
    sub[(size_t)(mBase + mo + r) * K_DIM + (nBase + nl)] = (int8_t)iv;
  }
}

__global__ __launch_bounds__(256) void gemm_sub(
    const _Float16* __restrict__ E, int8_t* __restrict__ sub) {
  const int lane = threadIdx.x & 31;
  const int gw = blockIdx.x * 8 + (threadIdx.x >> 5);  // global wave id
  const int tm = gw >> 7;    // 128 tiles per dim (4096/32)
  const int tn = gw & 127;
  const int mBase = tm * 32;
  const int nBase = tn * 32;

  v8f c00 = {}, c01 = {}, c10 = {}, c11 = {};

  for (int kb = 0; kb < D_DIM; kb += 32) {
    v16h a0 = load_a_frag(E, mBase,      kb, lane);
    v16h a1 = load_a_frag(E, mBase + 16, kb, lane);
    v16h b0 = load_b_frag(E, nBase,      kb, lane);
    v16h b1 = load_b_frag(E, nBase + 16, kb, lane);
    c00 = __builtin_amdgcn_wmma_f32_16x16x32_f16(false, a0, false, b0,
                                                 (short)0, c00, false, false);
    c01 = __builtin_amdgcn_wmma_f32_16x16x32_f16(false, a0, false, b1,
                                                 (short)0, c01, false, false);
    c10 = __builtin_amdgcn_wmma_f32_16x16x32_f16(false, a1, false, b0,
                                                 (short)0, c10, false, false);
    c11 = __builtin_amdgcn_wmma_f32_16x16x32_f16(false, a1, false, b1,
                                                 (short)0, c11, false, false);
  }

  const int nl = lane & 15;
  const int mo = (lane >> 4) << 3;  // 0 or 8
  store_tile(sub, c00, mBase,      nBase,      mo, nl);
  store_tile(sub, c01, mBase,      nBase + 16, mo, nl);
  store_tile(sub, c10, mBase + 16, nBase,      mo, nl);
  store_tile(sub, c11, mBase + 16, nBase + 16, mo, nl);
}

// ---------------------------------------------------------------------------
// Kernel 3: Needleman-Wunsch with linear gap, row recurrence resolved via
// block-wide inclusive max-scan (mirrors the reference's cummax trick).
// One 1024-thread block per batch; thread j owns column j+1.
// ---------------------------------------------------------------------------
__global__ __launch_bounds__(1024) void nw_score(
    const int* __restrict__ A, const int* __restrict__ Bids,
    const int8_t* __restrict__ sub, float* __restrict__ out) {
  const int batch = blockIdx.x;
  const int tid = threadIdx.x;
  const int lane = tid & 31;
  const int wv = tid >> 5;

  __shared__ int H[L_DIM + 1];
  __shared__ int wmax[32];

  const int bj = Bids[(size_t)batch * L_DIM + tid];
  const int* __restrict__ arow = A + (size_t)batch * L_DIM;

  // H[0, j] = GAP * j (terminal gaps penalized)
  H[tid] = GAP * tid;
  if (tid == 0) H[L_DIM] = GAP * L_DIM;
  __syncthreads();

  for (int i = 1; i <= L_DIM; ++i) {
    const int ai = arow[i - 1];  // uniform -> scalar load
    // Warm L2 path for next row's gathers while we do this row's scan.
    if (i < L_DIM) {
      const int an = arow[i];
      __builtin_prefetch(&sub[(size_t)an * K_DIM + bj], 0, 1);
    }
    const int s = (int)sub[(size_t)ai * K_DIM + bj];  // int8 sign-extend
    const int pjm1 = H[tid];      // H[i-1, j-1]
    const int pj   = H[tid + 1];  // H[i-1, j]
    const int T = max(pjm1 + s, pj + GAP);
    int v = T - GAP * (tid + 1);  // U[j] - GAP*j, j = tid+1

    // Inclusive max-scan across 1024 threads.
#pragma unroll
    for (int d = 1; d < 32; d <<= 1) {
      int t = __shfl_up(v, d, 32);
      if (lane >= d) v = max(v, t);
    }
    if (lane == 31) wmax[wv] = v;
    __syncthreads();  // also guarantees all H reads above are done
    if (wv == 0) {
      int w = wmax[lane];
#pragma unroll
      for (int d = 1; d < 32; d <<= 1) {
        int t = __shfl_up(w, d, 32);
        if (lane >= d) w = max(w, t);
      }
      wmax[lane] = w;
    }
    __syncthreads();
    if (wv > 0) v = max(v, wmax[wv - 1]);

    const int Hi0 = GAP * i;  // U[0] - GAP*0 = H[i,0]
    const int nv = max(v, Hi0) + GAP * (tid + 1);
    H[tid + 1] = nv;
    if (tid == 0) H[0] = Hi0;
    __syncthreads();
  }

  if (tid == L_DIM - 1) out[batch] = (float)H[L_DIM];
}

// ---------------------------------------------------------------------------
// Launch
// ---------------------------------------------------------------------------
extern "C" void kernel_launch(void* const* d_in, const int* in_sizes, int n_in,
                              void* d_out, int out_size, void* d_ws, size_t ws_size,
                              hipStream_t stream) {
  const int*   a     = (const int*)d_in[0];    // prot1_input_ids (32,1024)
  const int*   b     = (const int*)d_in[1];    // prot2_input_ids (32,1024)
  // d_in[2] = targets (unused by the reference output)
  const float* embed = (const float*)d_in[3];  // codebook_embedding (4096,512)
  float* out = (float*)d_out;

  // Workspace layout: [0, 16MB) int8 sub matrix; [16MB, 20MB) fp16 E.
  int8_t*   sub = (int8_t*)d_ws;
  _Float16* E   = (_Float16*)((char*)d_ws + (size_t)K_DIM * K_DIM);

  normalize_embed<<<K_DIM, 256, 0, stream>>>(embed, E);
  gemm_sub<<<(128 * 128) / 8, 256, 0, stream>>>(E, sub);
  nw_score<<<B_DIM, L_DIM, 0, stream>>>(a, b, sub, out);
}
